// double_resnet_23433341567710
// MI455X (gfx1250) — matrix operand order repaired and checked by reading, not compiled, not run on hardware
//
#include <hip/hip_runtime.h>
#include <hip/hip_bf16.h>

typedef __attribute__((ext_vector_type(16))) _Float16 v16h;
typedef __attribute__((ext_vector_type(8)))  _Float16 v8h;
typedef __attribute__((ext_vector_type(8)))  float    v8f;

#define HWP   65536      // 256*256
#define NB    4          // batch
#define BHW   262144     // NB*HWP
#define HALO  258
#define NL    5

// ---------------------------------------------------------------------------
// utility
// ---------------------------------------------------------------------------
__device__ __forceinline__ int clampi(int v, int lo, int hi) {
    return v < lo ? lo : (v > hi ? hi : v);
}

__device__ __forceinline__ float bilin256(const float* __restrict__ im, float x, float y) {
    float x0 = floorf(x), y0 = floorf(y);
    float wx = x - x0, wy = y - y0;
    float r = 0.f;
#pragma unroll
    for (int dy = 0; dy < 2; ++dy) {
#pragma unroll
        for (int dx = 0; dx < 2; ++dx) {
            float xi = x0 + (float)dx, yi = y0 + (float)dy;
            float w  = (dx ? wx : 1.f - wx) * (dy ? wy : 1.f - wy);
            float valid = (xi >= 0.f && xi <= 255.f && yi >= 0.f && yi <= 255.f) ? 1.f : 0.f;
            float xcf = fminf(fmaxf(xi, 0.f), 255.f);
            float ycf = fminf(fmaxf(yi, 0.f), 255.f);
            int xc = (int)xcf, yc = (int)ycf;
            r += im[yc * 256 + xc] * w * valid;
        }
    }
    return r;
}

// ---------------------------------------------------------------------------
// zero-fill (act buffers: halos + padded channels must be 0 each call)
// ---------------------------------------------------------------------------
__global__ void k_zero(unsigned int* __restrict__ p, size_t n) {
    size_t i = (size_t)blockIdx.x * blockDim.x + threadIdx.x;
    if (i < n) p[i] = 0u;
}

// ---------------------------------------------------------------------------
// weight packing into WMMA A-fragment lane order
// layout: [tap][chunk][tile][lane(32)][16 halfs], stride 512 halfs per frag
// A 16x32 f16, k order used here (B loads match): lane m(<16): halves0..15 -> K 0..15
//                     lane m+16 :  halves0..15 -> K16..31
// ---------------------------------------------------------------------------
__global__ void k_pack(const float* __restrict__ w, _Float16* __restrict__ dst,
                       int OC, int IC, int chunks, int tiles, int n) {
    int idx = blockIdx.x * blockDim.x + threadIdx.x;
    if (idx >= n) return;
    int j    = idx & 15;
    int lane = (idx >> 4) & 31;
    int frag = idx >> 9;
    int tile = frag % tiles;
    int q    = frag / tiles;
    int chunk = q % chunks;
    int tap   = q / chunks;
    int oc = tile * 16 + (lane & 15);
    int ic = chunk * 32 + ((lane & 16) ? 16 : 0) + j;
    float v = 0.f;
    if (oc < OC && ic < IC) {
        int ky = tap / 3, kx = tap - ky * 3;
        v = w[((size_t)(oc * IC + ic) * 3 + ky) * 3 + kx];
    }
    dst[idx] = (_Float16)v;
}

// ---------------------------------------------------------------------------
// implicit-GEMM 3x3 conv via v_wmma_f32_16x16x32_f16
// in : f16 NHWC (CS channels/pixel, compile-time) with 1-pixel zero halo,
//      dims NB x 258 x 258 x CS.  All fragment addresses are base + immediate.
// MODE 0: leaky-relu, store f16 into 128-ch halo buffer (out128)
// MODE 1: OC=1:  fout[p] = acc*scaleA + addin[p]     (residual update, f32)
// MODE 2: OC=2:  fout[(b*2+c)*HWP+p] = acc           (velocity head, f32)
// ---------------------------------------------------------------------------
template <int CS, int CHUNKS, int TILES, int MODE>
__global__ __launch_bounds__(256) void k_conv(const _Float16* __restrict__ in,
                                              const _Float16* __restrict__ wpk,
                                              _Float16* __restrict__ out128,
                                              const float* __restrict__ addin,
                                              float* __restrict__ fout, float scaleA) {
    int wid  = blockIdx.x * 8 + (threadIdx.x >> 5);
    int lane = threadIdx.x & 31;
    int b    = wid >> 12;          // 4096 waves per image (256 rows x 16 tiles)
    int rem  = wid & 4095;
    int y    = rem >> 4;
    int x0   = (rem & 15) << 4;
    int col  = lane & 15;
    int khalf = (lane & 16);       // 0 or 16 halfs: K-half select for B fragment

    // one base pointer per wave; every fragment address is an immediate offset
    const _Float16* pin = in + ((size_t)b * HALO * HALO
                                + (size_t)y * HALO + (size_t)(x0 + col)) * CS + khalf;
    const _Float16* wl  = wpk + lane * 16;

    v8f acc[TILES];
    v8f vzero = {0.f, 0.f, 0.f, 0.f, 0.f, 0.f, 0.f, 0.f};
#pragma unroll
    for (int t = 0; t < TILES; ++t) acc[t] = vzero;

#pragma unroll
    for (int tap = 0; tap < 9; ++tap) {
        const int ky = tap / 3, kx = tap - ky * 3;
        const int boff = (ky * HALO + kx) * CS;             // compile-time constant
#pragma unroll
        for (int c = 0; c < CHUNKS; ++c) {
            v16h bf = *(const v16h*)(pin + boff + c * 32);
            // grouped A-fragment loads so the scheduler can clause them ahead
            v16h af[TILES];
#pragma unroll
            for (int t = 0; t < TILES; ++t)
                af[t] = *(const v16h*)(wl + (size_t)((tap * CHUNKS + c) * TILES + t) * 512);
#pragma unroll
            for (int t = 0; t < TILES; ++t)
                acc[t] = __builtin_amdgcn_wmma_f32_16x16x32_f16(
                    false, af[t], false, bf, (short)0, acc[t], false, false);
        }
    }

    if (MODE == 0) {
        int choff = (lane & 16) ? 8 : 0;
        size_t opix = (((size_t)b * HALO + (y + 1)) * HALO + (x0 + 1 + col)) * 128;
#pragma unroll
        for (int t = 0; t < TILES; ++t) {
            v8h h;
#pragma unroll
            for (int g = 0; g < 8; ++g) {
                float v = acc[t][g];
                v = (v >= 0.f) ? v : 0.01f * v;
                h[g] = (_Float16)v;
            }
            *(v8h*)(out128 + opix + t * 16 + choff) = h;
        }
    } else if (MODE == 1) {
        if (lane < 16) {
            int p = b * HWP + y * 256 + x0 + lane;
            fout[p] = acc[0][0] * scaleA + addin[p];
        }
    } else {  // MODE 2
        if (lane < 16) {
            int p2 = y * 256 + x0 + lane;
            fout[(size_t)(b * 2 + 0) * HWP + p2] = acc[0][0];
            fout[(size_t)(b * 2 + 1) * HWP + p2] = acc[0][1];
        }
    }
}

// ---------------------------------------------------------------------------
// build f16 conv input (3 real channels of a 32-ch halo buffer)
// ---------------------------------------------------------------------------
__global__ void k_build_in(_Float16* __restrict__ act,
                           const float* __restrict__ c0, int s0,
                           const float* __restrict__ c1, int s1,
                           const float* __restrict__ c2, int s2) {
    int idx = blockIdx.x * blockDim.x + threadIdx.x;
    if (idx >= BHW) return;
    int b = idx >> 16, p2 = idx & 65535;
    int y = p2 >> 8, x = p2 & 255;
    size_t o = (((size_t)b * HALO + (y + 1)) * HALO + (x + 1)) * 32;
    act[o + 0] = (_Float16)c0[(size_t)b * s0 + p2];
    act[o + 1] = (_Float16)c1[(size_t)b * s1 + p2];
    act[o + 2] = (_Float16)c2[(size_t)b * s2 + p2];
}

// ---------------------------------------------------------------------------
// init: images[0]=src, residuals[0]=rd[0]=broadcast z0, deformation=identity
// ---------------------------------------------------------------------------
__global__ void k_init(const float* __restrict__ src, const float* __restrict__ z0,
                       float* __restrict__ img0, float* __restrict__ res0,
                       float* __restrict__ rd0, float* __restrict__ def) {
    int idx = blockIdx.x * blockDim.x + threadIdx.x;
    if (idx >= BHW) return;
    int b = idx >> 16, p2 = idx & 65535;
    int x = p2 & 255, y = p2 >> 8;
    img0[idx] = src[idx];
    float z = z0[p2];
    res0[idx] = z;
    rd0[idx]  = z;
    def[(size_t)(2 * b + 0) * HWP + p2] = (float)x;
    def[(size_t)(2 * b + 1) * HWP + p2] = (float)y;
}

// ---------------------------------------------------------------------------
// gaussian weights (51 taps, sigma=10, normalized)
// ---------------------------------------------------------------------------
__global__ void k_gauss(float* __restrict__ g) {
    if (threadIdx.x == 0 && blockIdx.x == 0) {
        float t[51];
        float s = 0.f;
        for (int k = 0; k < 51; ++k) {
            float xx = ((float)k - 25.f) * 0.1f;
            t[k] = expf(-0.5f * xx * xx);
            s += t[k];
        }
        for (int k = 0; k < 51; ++k) g[k] = t[k] / s;
    }
}

// ---------------------------------------------------------------------------
// separable 51-tap gaussian blur (zero pad). Vertical pass fuses field output
// and deformation update: def -= field/L
// ---------------------------------------------------------------------------
__global__ void k_blur_h(const float* __restrict__ in, float* __restrict__ out,
                         const float* __restrict__ g) {
    int n = blockIdx.x * blockDim.x + threadIdx.x;
    if (n >= 2 * BHW) return;
    int x = n & 255;
    int base = n - x;
    float s = 0.f;
    for (int k = 0; k < 51; ++k) {
        int xx = x + k - 25;
        if (xx >= 0 && xx < 256) s += g[k] * in[base + xx];
    }
    out[n] = s;
}

__global__ void k_blur_v(const float* __restrict__ in, float* __restrict__ field,
                         float* __restrict__ def, const float* __restrict__ g, float invL) {
    int n = blockIdx.x * blockDim.x + threadIdx.x;
    if (n >= 2 * BHW) return;
    int x = n & 255, y = (n >> 8) & 255, bc = n >> 16;
    float s = 0.f;
    for (int k = 0; k < 51; ++k) {
        int yy = y + k - 25;
        if (yy >= 0 && yy < 256) s += g[k] * in[(bc << 16) + (yy << 8) + x];
    }
    field[n] = s;
    def[n]  -= s * invL;
}

// ---------------------------------------------------------------------------
// deform residuals_deformed list (old deformation) + append residuals[i+1]
// ---------------------------------------------------------------------------
__global__ void k_deform_rd(const float* __restrict__ def, const float* __restrict__ rdA,
                            float* __restrict__ rdB, int count, const float* __restrict__ resn) {
    int idx = blockIdx.x * blockDim.x + threadIdx.x;
    if (idx >= BHW) return;
    int b = idx >> 16, p2 = idx & 65535;
    float xs = def[(size_t)(2 * b + 0) * HWP + p2];
    float ys = def[(size_t)(2 * b + 1) * HWP + p2];
    for (int j = 0; j < count; ++j)
        rdB[((size_t)j * NB + b) * HWP + p2] =
            bilin256(rdA + ((size_t)j * NB + b) * HWP, xs, ys);
    rdB[((size_t)count * NB + b) * HWP + p2] = resn[(size_t)b * HWP + p2];
}

// ---------------------------------------------------------------------------
// new_img = warp(img0, def) + (sum rd[1..last]) * coef * warp(seg, def)
// ---------------------------------------------------------------------------
__global__ void k_compose(const float* __restrict__ def, const float* __restrict__ seg,
                          const float* __restrict__ img0, const float* __restrict__ rd,
                          int last, float coef, float* __restrict__ imgout) {
    int idx = blockIdx.x * blockDim.x + threadIdx.x;
    if (idx >= BHW) return;
    int b = idx >> 16, p2 = idx & 65535;
    float xs = def[(size_t)(2 * b + 0) * HWP + p2];
    float ys = def[(size_t)(2 * b + 1) * HWP + p2];
    float mask = bilin256(seg  + (size_t)b * HWP, xs, ys);
    float base = bilin256(img0 + (size_t)b * HWP, xs, ys);
    float acc = 0.f;
    for (int j = 1; j <= last; ++j) acc += rd[((size_t)j * NB + b) * HWP + p2];
    imgout[(size_t)b * HWP + p2] = base + acc * coef * mask;
}

// ---------------------------------------------------------------------------
// spatial gradient (Sobel/8, edge-replicate pad), out channels: d/dx then d/dy
// ---------------------------------------------------------------------------
__global__ void k_sobel(const float* __restrict__ img, float* __restrict__ gout) {
    int idx = blockIdx.x * blockDim.x + threadIdx.x;
    if (idx >= BHW) return;
    int b = idx >> 16, p2 = idx & 65535;
    int y = p2 >> 8, x = p2 & 255;
    const float* im = img + (size_t)b * HWP;
    int ym = clampi(y - 1, 0, 255), yp = clampi(y + 1, 0, 255);
    int xm = clampi(x - 1, 0, 255), xp = clampi(x + 1, 0, 255);
    float a00 = im[ym * 256 + xm], a01 = im[ym * 256 + x], a02 = im[ym * 256 + xp];
    float a10 = im[y  * 256 + xm],                          a12 = im[y  * 256 + xp];
    float a20 = im[yp * 256 + xm], a21 = im[yp * 256 + x], a22 = im[yp * 256 + xp];
    float gx = (-a00 + a02 - 2.f * a10 + 2.f * a12 - a20 + a22) * 0.125f;
    float gy = (-a00 - 2.f * a01 - a02 + a20 + 2.f * a21 + a22) * 0.125f;
    gout[(size_t)(2 * b + 0) * HWP + p2] = gx;
    gout[(size_t)(2 * b + 1) * HWP + p2] = gy;
}

// ---------------------------------------------------------------------------
// host
// ---------------------------------------------------------------------------
extern "C" void kernel_launch(void* const* d_in, const int* in_sizes, int n_in,
                              void* d_out, int out_size, void* d_ws, size_t ws_size,
                              hipStream_t stream) {
    const float* SRC = (const float*)d_in[0];
    const float* SEG = (const float*)d_in[1];
    const float* Z0  = (const float*)d_in[2];
    const float* ZW1 = (const float*)d_in[3];
    const float* ZW2 = (const float*)d_in[4];
    const float* ZW3 = (const float*)d_in[5];
    const float* VW1 = (const float*)d_in[6];
    const float* VW2 = (const float*)d_in[7];
    const float* VW3 = (const float*)d_in[8];

    float* OUT = (float*)d_out;
    float* IMG = OUT;                       // (6,4,1,256,256)
    float* FLD = OUT + 1572864;             // (5,4,2,256,256)
    float* RES = OUT + 4194304;             // (6,4,1,256,256)
    float* GRD = OUT + 5767168;             // (6,4,1,2,256,256)

    // workspace layout (bytes)
    char* W = (char*)d_ws;
    const size_t szActIn = (size_t)NB * HALO * HALO * 32 * 2;    // 17,040,384
    const size_t szActH  = (size_t)NB * HALO * HALO * 128 * 2;   // 68,161,536
    const size_t oActIn = 0;
    const size_t oActH1 = oActIn + szActIn;
    const size_t oActH2 = oActH1 + szActH;
    const size_t oPW    = oActH2 + szActH;                       // 153,363,456
    const size_t szPW   = (size_t)NL * 359424 * 2;
    const size_t oRdA   = oPW + szPW;
    const size_t oRdB   = oRdA + 6ull * BHW * 4;
    const size_t oDef   = oRdB + 6ull * BHW * 4;
    const size_t oVt1   = oDef + 2ull * BHW * 4;
    const size_t oVt2   = oVt1 + 2ull * BHW * 4;
    const size_t oG     = oVt2 + 2ull * BHW * 4;

    _Float16* actIn = (_Float16*)(W + oActIn);
    _Float16* actH1 = (_Float16*)(W + oActH1);
    _Float16* actH2 = (_Float16*)(W + oActH2);
    _Float16* pw    = (_Float16*)(W + oPW);
    float* rdA = (float*)(W + oRdA);
    float* rdB = (float*)(W + oRdB);
    float* def = (float*)(W + oDef);
    float* vt1 = (float*)(W + oVt1);
    float* vt2 = (float*)(W + oVt2);
    float* gw  = (float*)(W + oG);

    // zero all f16 activation buffers (halos + padded channels)
    {
        size_t ndw = (szActIn + 2 * szActH) / 4;
        int blocks = (int)((ndw + 255) / 256);
        k_zero<<<blocks, 256, 0, stream>>>((unsigned int*)W, ndw);
    }

    // pack weights for all 5 layers, 6 convs each
    for (int l = 0; l < NL; ++l) {
        _Float16* base = pw + (size_t)l * 359424;
        struct { const float* w; size_t doff; int OC, IC, ch, ti; } cfg[6] = {
            { ZW1 + (size_t)l * 2700,  0,      100, 3,   1, 7 },
            { ZW2 + (size_t)l * 90000, 32256,  100, 100, 4, 7 },
            { ZW3 + (size_t)l * 900,   161280, 1,   100, 4, 1 },
            { VW1 + (size_t)l * 2700,  179712, 100, 3,   1, 7 },
            { VW2 + (size_t)l * 90000, 211968, 100, 100, 4, 7 },
            { VW3 + (size_t)l * 1800,  340992, 2,   100, 4, 1 },
        };
        for (int c = 0; c < 6; ++c) {
            int n = 9 * cfg[c].ch * cfg[c].ti * 512;
            k_pack<<<(n + 255) / 256, 256, 0, stream>>>(cfg[c].w, base + cfg[c].doff,
                                                        cfg[c].OC, cfg[c].IC,
                                                        cfg[c].ch, cfg[c].ti, n);
        }
    }

    k_gauss<<<1, 64, 0, stream>>>(gw);
    k_init<<<BHW / 256, 256, 0, stream>>>(SRC, Z0, IMG, RES, rdA, def);
    k_sobel<<<BHW / 256, 256, 0, stream>>>(SRC, GRD);

    const float invL = 1.f / (float)NL;
    const float coef = 0.02f * 0.02f / (float)NL;   // MU^2 / L

    for (int i = 0; i < NL; ++i) {
        const _Float16* wb = pw + (size_t)i * 359424;
        const float* img_i = IMG + (size_t)i * BHW;
        const float* res_i = RES + (size_t)i * BHW;
        float* res_n = RES + (size_t)(i + 1) * BHW;

        // --- residual branch: zin = [residual_i, image_i, image_0] ---
        k_build_in<<<BHW / 256, 256, 0, stream>>>(actIn, res_i, HWP, img_i, HWP, IMG, HWP);
        k_conv<32, 1, 7, 0><<<2048, 256, 0, stream>>>(actIn, wb + 0,      actH1, nullptr, nullptr, 0.f);
        k_conv<128, 4, 7, 0><<<2048, 256, 0, stream>>>(actH1, wb + 32256,  actH2, nullptr, nullptr, 0.f);
        k_conv<128, 4, 1, 1><<<2048, 256, 0, stream>>>(actH2, wb + 161280, nullptr, res_i, res_n, invL);

        // --- deform residual list with OLD deformation, append new residual ---
        k_deform_rd<<<BHW / 256, 256, 0, stream>>>(def, rdA, rdB, i + 1, res_n);
        { float* t = rdA; rdA = rdB; rdB = t; }

        // --- velocity branch: vin = [def_x, def_y, image_i] ---
        k_build_in<<<BHW / 256, 256, 0, stream>>>(actIn, def, 2 * HWP, def + HWP, 2 * HWP, img_i, HWP);
        k_conv<32, 1, 7, 0><<<2048, 256, 0, stream>>>(actIn, wb + 179712, actH1, nullptr, nullptr, 0.f);
        k_conv<128, 4, 7, 0><<<2048, 256, 0, stream>>>(actH1, wb + 211968, actH2, nullptr, nullptr, 0.f);
        k_conv<128, 4, 1, 2><<<2048, 256, 0, stream>>>(actH2, wb + 340992, nullptr, nullptr, vt1, 1.f);

        // --- gaussian blur -> field; def -= field/L (fused in vertical pass) ---
        k_blur_h<<<2 * BHW / 256, 256, 0, stream>>>(vt1, vt2, gw);
        k_blur_v<<<2 * BHW / 256, 256, 0, stream>>>(vt2, FLD + (size_t)i * 2 * BHW, def, gw, invL);

        // --- compose new image with NEW deformation, then its gradient ---
        k_compose<<<BHW / 256, 256, 0, stream>>>(def, SEG, IMG, rdA, i + 1, coef,
                                                 IMG + (size_t)(i + 1) * BHW);
        k_sobel<<<BHW / 256, 256, 0, stream>>>(IMG + (size_t)(i + 1) * BHW,
                                               GRD + (size_t)(i + 1) * 2 * BHW);
    }

    (void)in_sizes; (void)n_in; (void)out_size; (void)ws_size;
}
